// FCGTransformerEncoder_3659312136717
// MI455X (gfx1250) — compile-verified, matrix operand order checked by reading, not run on hardware
//
#include <hip/hip_runtime.h>
#include <hip/hip_fp16.h>

// ---------------------------------------------------------------------------
// Focus-DETR encoder for MI455X (gfx1250): wave32 + V_WMMA_F32_16X16X32_F16
// All WMMA operands are stored K-contiguous so every fragment is 2x b128 loads.
// GEMM uses 32x64 register blocking per wave: 8 WMMAs per 6 fragment loads.
// ---------------------------------------------------------------------------

typedef __attribute__((ext_vector_type(16))) _Float16 v16h;
typedef __attribute__((ext_vector_type(8)))  _Float16 v8h;
typedef __attribute__((ext_vector_type(8)))  float    v8f;
typedef _Float16 half_t;

#define LAYERS 6
#define Bb     4
#define Ntok   23890
#define KTOK   1024
#define Dm     256
#define Hh     8
#define HD     32
#define FFNm   1024
#define NCLS   15
#define BK     (Bb * KTOK)   // 4096 rows processed jointly (weights batch-shared)
#define WM     2             // 16-row tiles per wave
#define WN     4             // 16-col tiles per wave

// ---------------------------- wave32 reductions ----------------------------
__device__ inline float wave_sum(float v) {
#pragma unroll
  for (int m = 16; m >= 1; m >>= 1) v += __shfl_xor(v, m, 32);
  return v;
}

// ------------------------------ WMMA helpers -------------------------------
__device__ inline v8f wmma16(v16h a, v16h b, v8f c) {
  // D = A(16x32 f16) * B(32x16 f16) + C(16x16 f32)
  return __builtin_amdgcn_wmma_f32_16x16x32_f16(
      /*neg_a=*/false, a, /*neg_b=*/false, b,
      /*c_mod=*/(short)0, c, /*reuse_a=*/false, /*reuse_b=*/false);
}

// Fragment from a K-contiguous (row-major [rows, ld]) matrix.
// Lane r = lane&15 picks the row; K halves {kb..kb+7, kb+16..kb+23},
// kb = k0 + 8*(lane>=16). Two aligned 16B loads, merged with shufflevector.
__device__ inline v16h load_frag(const half_t* __restrict__ base, int ld,
                                 int r0, int k0, int lane) {
  const half_t* p = base + (size_t)(r0 + (lane & 15)) * ld + k0 + ((lane >> 4) << 3);
  v8h lo = *(const v8h*)p;
  v8h hi = *(const v8h*)(p + 16);
  return __builtin_shufflevector(lo, hi, 0, 1, 2, 3, 4, 5, 6, 7,
                                 8, 9, 10, 11, 12, 13, 14, 15);
}

// ---------------------- weight convert f32 -> f16 (transposed) --------------
// src row-major [K,N] -> dst row-major [N,K]  (so B-fragments are K-contiguous)
__global__ void k_cvt_T(const float* __restrict__ src, half_t* __restrict__ dst,
                        int K, int N) {
  int i = blockIdx.x * blockDim.x + threadIdx.x;
  if (i < K * N) {
    int k = i / N, n = i - k * N;
    dst[(size_t)n * K + k] = (half_t)src[i];
  }
}

// ---------------- gather + MCSP score modulation (wave per row) -------------
// tgt = q * (sigmoid(max_c(q@Wcls + bcls)) * fg);  qk = tgt + query_pos
__global__ void k_gather_mod(const float* __restrict__ state,      // [B,N,D] (d_out)
                             const float* __restrict__ query_pos,  // [B,N,D]
                             const float* __restrict__ fg_score,   // [B,N]
                             const int*   __restrict__ inds,       // [B,K] (layer slice)
                             const float* __restrict__ Wcls,       // [D,NCLS]
                             const float* __restrict__ bcls,       // [NCLS]
                             float*  __restrict__ tgt32,           // [BK,D]
                             half_t* __restrict__ tgt16,           // [BK,D]
                             half_t* __restrict__ qk16)            // [BK,D]
{
  int row  = blockIdx.x * (blockDim.x >> 5) + (threadIdx.x >> 5);
  int lane = threadIdx.x & 31;
  if (row >= BK) return;
  int b = row / KTOK, k = row - b * KTOK;
  int idx = inds[b * KTOK + k];
  const float* qp = state + ((size_t)b * Ntok + idx) * Dm;
  float q[8];
#pragma unroll
  for (int i = 0; i < 8; i++) q[i] = qp[lane + 32 * i];
  // scores over 15 classes; sigmoid monotone => max(sigmoid(s)) = sigmoid(max s)
  float best = -3.0e38f;
  for (int c = 0; c < NCLS; c++) {
    float s = 0.f;
#pragma unroll
    for (int i = 0; i < 8; i++) s += q[i] * Wcls[(lane + 32 * i) * NCLS + c];
    s = wave_sum(s) + bcls[c];
    best = fmaxf(best, s);
  }
  float mc = (1.f / (1.f + __expf(-best))) * fg_score[(size_t)b * Ntok + idx];
  const float* pp = query_pos + ((size_t)b * Ntok + idx) * Dm;
  float*  t32 = tgt32 + (size_t)row * Dm;
  half_t* t16 = tgt16 + (size_t)row * Dm;
  half_t* qk  = qk16  + (size_t)row * Dm;
#pragma unroll
  for (int i = 0; i < 8; i++) {
    int d = lane + 32 * i;
    float t = q[i] * mc;
    t32[d] = t;
    t16[d] = (half_t)t;
    qk[d]  = (half_t)(t + pp[d]);
  }
}

// ----------------------- generic WMMA GEMM (f16 in) -------------------------
// C[M,N] = A[M,Kd] @ Bt^T + bias, with Bt stored [N,Kd] (K-contiguous).
// Wave computes a 32x64 block (2x4 tiles of 16x16) -> 8 WMMAs per k-step
// against 6 fragment loads. mode 0: f32 out ; 1: f16 out ; 2: f16 relu ;
// 3: f16 transposed-per-head out (out16 = Vt[B][H][HD][KTOK]).
__global__ void k_gemm_wmma(const half_t* __restrict__ A, const half_t* __restrict__ Bt,
                            const float* __restrict__ bias,
                            float* __restrict__ out32, half_t* __restrict__ out16,
                            int M, int N, int Kd, int mode)
{
  int wave = blockIdx.x * (blockDim.x >> 5) + (threadIdx.x >> 5);
  int lane = threadIdx.x & 31;
  int nb = N >> 6;                          // 64-col blocks per row
  int wm = wave / nb, wn = wave - wm * nb;
  if (wm * (16 * WM) >= M) return;          // wave-uniform exit (EXEC stays all-1)
  int m0 = wm * (16 * WM), n0 = wn * (16 * WN);

  v8f acc[WM][WN];
#pragma unroll
  for (int in = 0; in < WN; in++) {
    float bv = bias ? bias[n0 + 16 * in + (lane & 15)] : 0.f;
#pragma unroll
    for (int im = 0; im < WM; im++)
#pragma unroll
      for (int i = 0; i < 8; i++) acc[im][in][i] = bv;
  }

  for (int k0 = 0; k0 < Kd; k0 += 32) {
    v16h a[WM], b[WN];
#pragma unroll
    for (int im = 0; im < WM; im++) a[im] = load_frag(A, Kd, m0 + 16 * im, k0, lane);
#pragma unroll
    for (int in = 0; in < WN; in++) b[in] = load_frag(Bt, Kd, n0 + 16 * in, k0, lane);
    // prefetch next k-step (speculative; dropped if past end)
    __builtin_prefetch(A  + (size_t)(m0 + (lane & 15)) * Kd + k0 + 32, 0, 3);
    __builtin_prefetch(Bt + (size_t)(n0 + (lane & 15)) * Kd + k0 + 32, 0, 3);
#pragma unroll
    for (int im = 0; im < WM; im++)
#pragma unroll
      for (int in = 0; in < WN; in++)
        acc[im][in] = wmma16(a[im], b[in], acc[im][in]);
  }

#pragma unroll
  for (int im = 0; im < WM; im++) {
#pragma unroll
    for (int in = 0; in < WN; in++) {
      int col  = n0 + 16 * in + (lane & 15);
      int row0 = m0 + 16 * im + ((lane >> 4) << 3);
      if (mode == 0) {
#pragma unroll
        for (int i = 0; i < 8; i++) out32[(size_t)(row0 + i) * N + col] = acc[im][in][i];
      } else if (mode == 3) {
        int bset = row0 >> 10;              // batch (KTOK rows per batch)
        int kk   = row0 & (KTOK - 1);       // contiguous along K in Vt
        int h    = col >> 5, j = col & 31;
        v8h o;
#pragma unroll
        for (int i = 0; i < 8; i++) o[i] = (half_t)acc[im][in][i];
        *(v8h*)(out16 + (((size_t)(bset * Hh + h) * HD + j) * KTOK + kk)) = o;
      } else {
#pragma unroll
        for (int i = 0; i < 8; i++) {
          float v = acc[im][in][i];
          if (mode == 2) v = fmaxf(v, 0.f);
          out16[(size_t)(row0 + i) * N + col] = (half_t)v;
        }
      }
    }
  }
}

// --------------- fused attention per (b, h, 16-query block) -----------------
// S = (Q Kt^T)/sqrt(hd) in 64KB LDS, softmax in place, then O = P @ V (WMMA).
// K16 is row-major [b*K, D] (already K-contiguous for the S GEMM B-operand);
// Vt is [B][H][HD][KTOK] (K-contiguous for the PV GEMM B-operand).
__global__ __launch_bounds__(128)
void k_attention(const half_t* __restrict__ Q16, const half_t* __restrict__ K16,
                 const half_t* __restrict__ Vt, half_t* __restrict__ O16)
{
  __shared__ float S[16 * KTOK];            // exactly 64 KB
  int bid  = blockIdx.x;                    // b*512 + h*64 + qblk
  int qblk = bid & 63;
  int h    = (bid >> 6) & 7;
  int b    = bid >> 9;
  int lane = threadIdx.x & 31;
  int wv   = threadIdx.x >> 5;              // 0..3
  const half_t* Qh  = Q16 + (size_t)b * KTOK * Dm + h * HD;
  const half_t* Kh  = K16 + (size_t)b * KTOK * Dm + h * HD;
  const half_t* Vth = Vt  + ((size_t)(b * Hh + h) * HD) * KTOK;
  int q0 = qblk * 16;
  const float scale = 0.1767766953f;        // 1/sqrt(32)

  // A fragment (Q rows, full k=32 head) loaded once, reused for 16 WMMAs/wave
  v16h afrag = load_frag(Qh, Dm, q0, 0, lane);
  // Phase 1: 64 key tiles -> S
  for (int tn = wv; tn < 64; tn += 4) {
    int n0 = tn * 16;
    v16h bfrag = load_frag(Kh, Dm, n0, 0, lane);   // B[k][key]=Kh[key][k], contiguous
    v8f acc = {};
    acc = wmma16(afrag, bfrag, acc);
    int col = n0 + (lane & 15);
    int r0  = (lane >> 4) << 3;
#pragma unroll
    for (int i = 0; i < 8; i++) S[(r0 + i) * KTOK + col] = acc[i] * scale;
  }
  __syncthreads();
  // Phase 2: softmax; 8 threads per row for max/exp/sum (shuffle reduce),
  // then one thread per row does the safe ascending in-place f32->f16 pack.
  {
    int r = threadIdx.x >> 3;               // 0..15
    int t = threadIdx.x & 7;
    float* rowp = S + r * KTOK;
    const float4* rq = (const float4*)rowp;
    float mx = -3.0e38f;
    for (int i = t; i < KTOK / 4; i += 8) {
      float4 v = rq[i];
      mx = fmaxf(mx, fmaxf(fmaxf(v.x, v.y), fmaxf(v.z, v.w)));
    }
#pragma unroll
    for (int m = 4; m >= 1; m >>= 1) mx = fmaxf(mx, __shfl_xor(mx, m, 32));
    float sum = 0.f;
    float4* wq = (float4*)rowp;
    for (int i = t; i < KTOK / 4; i += 8) {
      float4 v = wq[i];
      v.x = __expf(v.x - mx); v.y = __expf(v.y - mx);
      v.z = __expf(v.z - mx); v.w = __expf(v.w - mx);
      wq[i] = v;
      sum += v.x + v.y + v.z + v.w;
    }
#pragma unroll
    for (int m = 4; m >= 1; m >>= 1) sum += __shfl_xor(sum, m, 32);
    if (t == 0) {                           // same wave: LDS writes already ordered
      float inv = 1.f / sum;
      half_t* ph = (half_t*)rowp;           // ascending in-place compaction is safe
      for (int c = 0; c < KTOK; c++) ph[c] = (half_t)(rowp[c] * inv);
    }
  }
  __syncthreads();
  // Phase 3: O = P(16x1024 f16 in LDS) @ V ; 2 output tiles (waves 0,1)
  if (wv < 2) {
    int n0 = wv * 16;
    v8f acc = {};
    const half_t* Plds = (const half_t*)S;  // rows of 2*KTOK halves (f16 in low half)
    for (int k0 = 0; k0 < KTOK; k0 += 32) {
      v16h a = load_frag(Plds, 2 * KTOK, 0, k0, lane);
      v16h bb = load_frag(Vth, KTOK, n0, k0, lane);
      acc = wmma16(a, bb, acc);
    }
    half_t* Oh = O16 + (size_t)b * KTOK * Dm + h * HD;
    int col = n0 + (lane & 15);
    int r0  = (lane >> 4) << 3;
#pragma unroll
    for (int i = 0; i < 8; i++) Oh[(size_t)(q0 + r0 + i) * Dm + col] = (half_t)acc[i];
  }
}

// --------------------- residual add + LayerNorm (wave/row) ------------------
__global__ void k_add_ln(const float* __restrict__ a, const float* __restrict__ r,
                         const float* __restrict__ g, const float* __restrict__ beta,
                         float* __restrict__ out32, half_t* __restrict__ out16)
{
  int row  = blockIdx.x * (blockDim.x >> 5) + (threadIdx.x >> 5);
  int lane = threadIdx.x & 31;
  if (row >= BK) return;
  const float* pa = a + (size_t)row * Dm;
  const float* pr = r + (size_t)row * Dm;
  float x[8]; float s = 0.f;
#pragma unroll
  for (int i = 0; i < 8; i++) { x[i] = pa[lane + 32 * i] + pr[lane + 32 * i]; s += x[i]; }
  float mu = wave_sum(s) * (1.f / Dm);
  float v = 0.f;
#pragma unroll
  for (int i = 0; i < 8; i++) { float d0 = x[i] - mu; v += d0 * d0; }
  float inv = rsqrtf(wave_sum(v) * (1.f / Dm) + 1e-5f);
#pragma unroll
  for (int i = 0; i < 8; i++) {
    int d = lane + 32 * i;
    float y = (x[i] - mu) * inv * g[d] + beta[d];
    out32[(size_t)row * Dm + d] = y;
    if (out16) out16[(size_t)row * Dm + d] = (half_t)y;
  }
}

// ------------------- ragged scatter back into state -------------------------
__global__ void k_scatter(const float* __restrict__ src, const int* __restrict__ inds,
                          const int* __restrict__ ftn, float* __restrict__ state)
{
  int row  = blockIdx.x * (blockDim.x >> 5) + (threadIdx.x >> 5);
  int lane = threadIdx.x & 31;
  if (row >= BK) return;
  int b = row / KTOK, k = row - b * KTOK;
  if (k >= ftn[b]) return;
  int idx = inds[b * KTOK + k];
  const float* s = src + (size_t)row * Dm;
  float* d0 = state + ((size_t)b * Ntok + idx) * Dm;
#pragma unroll
  for (int i = 0; i < 8; i++) d0[lane + 32 * i] = s[lane + 32 * i];
}

// ---------------------------------------------------------------------------
extern "C" void kernel_launch(void* const* d_in, const int* in_sizes, int n_in,
                              void* d_out, int out_size, void* d_ws, size_t ws_size,
                              hipStream_t stream) {
  (void)in_sizes; (void)n_in; (void)out_size; (void)ws_size;
  const float* query     = (const float*)d_in[0];
  const float* query_pos = (const float*)d_in[1];
  const float* fg_score  = (const float*)d_in[6];
  const int*   ftn       = (const int*)d_in[7];
  const int*   finds     = (const int*)d_in[8];
  const float* Wq = (const float*)d_in[9];  const float* bq = (const float*)d_in[10];
  const float* Wk = (const float*)d_in[11]; const float* bk = (const float*)d_in[12];
  const float* Wv = (const float*)d_in[13]; const float* bv = (const float*)d_in[14];
  const float* Wo = (const float*)d_in[15]; const float* bo = (const float*)d_in[16];
  const float* ln1g = (const float*)d_in[17]; const float* ln1b = (const float*)d_in[18];
  const float* W1 = (const float*)d_in[19]; const float* b1 = (const float*)d_in[20];
  const float* W2 = (const float*)d_in[21]; const float* b2 = (const float*)d_in[22];
  const float* ln2g = (const float*)d_in[23]; const float* ln2b = (const float*)d_in[24];
  const float* Wcls = (const float*)d_in[25]; const float* bcls = (const float*)d_in[26];
  float* state = (float*)d_out;                     // [B,N,D] evolving encoder state

  // ---- workspace carve (256B aligned) ----
  char* w = (char*)d_ws;
  auto carve = [&](size_t bytes) { char* p = w; w += (bytes + 255) & ~(size_t)255; return p; };
  float*  tgt32  = (float*)carve((size_t)BK * Dm * 4);
  float*  x32    = (float*)carve((size_t)BK * Dm * 4);
  float*  f32buf = (float*)carve((size_t)BK * Dm * 4);      // o-proj / ffn2 output
  float*  qnew32 = (float*)carve((size_t)BK * Dm * 4);
  half_t* tgt16  = (half_t*)carve((size_t)BK * Dm * 2);
  half_t* qk16   = (half_t*)carve((size_t)BK * Dm * 2);
  half_t* x16    = (half_t*)carve((size_t)BK * Dm * 2);
  half_t* Q16    = (half_t*)carve((size_t)BK * Dm * 2);
  half_t* K16v   = (half_t*)carve((size_t)BK * Dm * 2);
  half_t* Vt     = (half_t*)carve((size_t)BK * Dm * 2);     // [B][H][HD][KTOK]
  half_t* attn16 = (half_t*)carve((size_t)BK * Dm * 2);
  half_t* ffn1   = (half_t*)carve((size_t)BK * FFNm * 2);
  half_t* WqT    = (half_t*)carve((size_t)Dm * Dm * 2);     // all weights stored [N,Kd]
  half_t* WkT    = (half_t*)carve((size_t)Dm * Dm * 2);
  half_t* WvT    = (half_t*)carve((size_t)Dm * Dm * 2);
  half_t* WoT    = (half_t*)carve((size_t)Dm * Dm * 2);
  half_t* W1T    = (half_t*)carve((size_t)Dm * FFNm * 2);
  half_t* W2T    = (half_t*)carve((size_t)FFNm * Dm * 2);

  // state starts as query
  hipMemcpyAsync(state, query, (size_t)Bb * Ntok * Dm * sizeof(float),
                 hipMemcpyDeviceToDevice, stream);

  auto cvtT = [&](const float* s, half_t* d, int K, int N) {
    int n = K * N;
    k_cvt_T<<<(n + 255) / 256, 256, 0, stream>>>(s, d, K, N);
  };
  auto gemm = [&](const half_t* A, const half_t* Bt, const float* bias,
                  float* o32, half_t* o16, int M, int N, int Kd, int mode) {
    int tiles = (M / (16 * WM)) * (N / (16 * WN));   // 32x64 wave tiles
    k_gemm_wmma<<<tiles / 8, 256, 0, stream>>>(A, Bt, bias, o32, o16, M, N, Kd, mode);
  };

  const int ROWBLK = BK / 4;   // 4 waves (rows) per 128-thread block

  for (int l = 0; l < LAYERS; l++) {
    const int* il = finds + (size_t)l * Bb * KTOK;
    cvtT(Wq + (size_t)l * Dm * Dm, WqT, Dm, Dm);
    cvtT(Wk + (size_t)l * Dm * Dm, WkT, Dm, Dm);
    cvtT(Wv + (size_t)l * Dm * Dm, WvT, Dm, Dm);
    cvtT(Wo + (size_t)l * Dm * Dm, WoT, Dm, Dm);
    cvtT(W1 + (size_t)l * Dm * FFNm, W1T, Dm, FFNm);
    cvtT(W2 + (size_t)l * FFNm * Dm, W2T, FFNm, Dm);

    k_gather_mod<<<ROWBLK, 128, 0, stream>>>(state, query_pos, fg_score, il,
                                             Wcls, bcls, tgt32, tgt16, qk16);
    // Q/K from (tgt+pos), V from tgt (V written transposed per head)
    gemm(qk16,  WqT, bq + (size_t)l * Dm, nullptr, Q16,  BK, Dm, Dm, 1);
    gemm(qk16,  WkT, bk + (size_t)l * Dm, nullptr, K16v, BK, Dm, Dm, 1);
    gemm(tgt16, WvT, bv + (size_t)l * Dm, nullptr, Vt,   BK, Dm, Dm, 3);

    k_attention<<<Bb * Hh * (KTOK / 16), 128, 0, stream>>>(Q16, K16v, Vt, attn16);

    // o = attn @ Wo + bo  (f32 for residual)
    gemm(attn16, WoT, bo + (size_t)l * Dm, f32buf, nullptr, BK, Dm, Dm, 0);
    k_add_ln<<<ROWBLK, 128, 0, stream>>>(tgt32, f32buf, ln1g + (size_t)l * Dm,
                                         ln1b + (size_t)l * Dm, x32, x16);
    // FFN
    gemm(x16,  W1T, b1 + (size_t)l * FFNm, nullptr, ffn1, BK, FFNm, Dm, 2);   // relu
    gemm(ffn1, W2T, b2 + (size_t)l * Dm,   f32buf, nullptr, BK, Dm, FFNm, 0);
    k_add_ln<<<ROWBLK, 128, 0, stream>>>(x32, f32buf, ln2g + (size_t)l * Dm,
                                         ln2b + (size_t)l * Dm, qnew32, (half_t*)nullptr);

    k_scatter<<<ROWBLK, 128, 0, stream>>>(qnew32, il, ftn, state);
  }
}
